// WindowAttentionGlobal_65180423685131
// MI455X (gfx1250) — matrix operand order repaired and checked by reading, not compiled, not run on hardware
//
#include <hip/hip_runtime.h>
#include <hip/hip_fp16.h>

typedef __attribute__((ext_vector_type(16))) _Float16 v16h;
typedef __attribute__((ext_vector_type(8)))  _Float16 v8h;
typedef __attribute__((ext_vector_type(2)))  _Float16 v2h;
typedef __attribute__((ext_vector_type(8)))  float    v8f;
typedef __attribute__((ext_vector_type(4)))  unsigned int v4u;
typedef __attribute__((ext_vector_type(8)))  int v8i;
typedef __attribute__((ext_vector_type(4)))  int v4i;

#ifndef __has_builtin
#define __has_builtin(x) 0
#endif
#if __has_builtin(__builtin_amdgcn_tensor_load_to_lds) && __has_builtin(__builtin_amdgcn_s_wait_tensorcnt)
#define USE_TDM 1
#else
#define USE_TDM 0
#endif

// ---- problem constants ------------------------------------------------------
#define NWIN   4096        // B_
#define NTOK   49          // window tokens
#define CDIM   256
#define HEADS  8
#define HD     32
#define MROWS  (NWIN * NTOK)   // 200704 (exactly 1568 * 128)

__device__ __forceinline__ v16h ldsfrag(const _Float16* p0, const _Float16* p1) {
  union { v16h v; v8h h[2]; } u;
  u.h[0] = *(const v8h*)p0;
  u.h[1] = *(const v8h*)p1;
  return u.v;
}

__device__ __forceinline__ v8f wmma32(v16h a, v16h b, v8f c) {
  return __builtin_amdgcn_wmma_f32_16x16x32_f16(false, a, false, b, (short)0, c,
                                                false, false);
}

// ============================================================================
// Stage 1: kv = x @ qkv_w^T + qkv_b   -> f16 workspace [MROWS, 512]
// block tile 128(M) x 64(N), K loop of 32, 8 waves (2x2 wmma tiles per wave)
// ============================================================================
__global__ __launch_bounds__(256) void kv_gemm_f16(
    const float* __restrict__ x, const float* __restrict__ w,
    const float* __restrict__ bvec, _Float16* __restrict__ out) {
  __shared__ __align__(16) char smem[18432];
  _Float16* sA = (_Float16*)smem;            // 128 rows x 40 halfs (80B stride)
  _Float16* sB = (_Float16*)(smem + 10240);  // 64 rows x 40 halfs
  _Float16* sO = (_Float16*)smem;            // 128 rows x 72 halfs (bounce)

  const int tid = threadIdx.x, lane = tid & 31, wave = tid >> 5;
  const int wm = wave >> 1, wn = wave & 1;
  const int bm = blockIdx.x * 128, bn = blockIdx.y * 64;
  const int l15 = lane & 15, lhi = lane >> 4;

  v8f acc00 = {}, acc01 = {}, acc10 = {}, acc11 = {};

  const int ar = tid >> 1;          // A row loaded by this thread
  const int as = (tid & 1) * 16;    // half-offset within the 32-wide K chunk
  const int brn = tid >> 2;         // B row (output column)
  const int bsg = (tid & 3) * 8;

  for (int k0 = 0; k0 < CDIM; k0 += 32) {
    __syncthreads();
    {
      const float* fp = x + (size_t)(bm + ar) * CDIM + k0 + as;
      float4 a0 = ((const float4*)fp)[0];
      float4 a1 = ((const float4*)fp)[1];
      float4 a2 = ((const float4*)fp)[2];
      float4 a3 = ((const float4*)fp)[3];
      v8h h0 = {(_Float16)a0.x, (_Float16)a0.y, (_Float16)a0.z, (_Float16)a0.w,
                (_Float16)a1.x, (_Float16)a1.y, (_Float16)a1.z, (_Float16)a1.w};
      v8h h1 = {(_Float16)a2.x, (_Float16)a2.y, (_Float16)a2.z, (_Float16)a2.w,
                (_Float16)a3.x, (_Float16)a3.y, (_Float16)a3.z, (_Float16)a3.w};
      *(v8h*)(sA + ar * 40 + as) = h0;
      *(v8h*)(sA + ar * 40 + as + 8) = h1;

      const float* wp = w + (size_t)(bn + brn) * CDIM + k0 + bsg;
      float4 b0 = ((const float4*)wp)[0];
      float4 b1 = ((const float4*)wp)[1];
      v8h hb = {(_Float16)b0.x, (_Float16)b0.y, (_Float16)b0.z, (_Float16)b0.w,
                (_Float16)b1.x, (_Float16)b1.y, (_Float16)b1.z, (_Float16)b1.w};
      *(v8h*)(sB + brn * 40 + bsg) = hb;
    }
    __syncthreads();

    const int arow = wm * 32 + l15;
    const int ak = lhi * 8;
    v16h fa0 = ldsfrag(sA + arow * 40 + ak, sA + arow * 40 + ak + 16);
    v16h fa1 = ldsfrag(sA + (arow + 16) * 40 + ak, sA + (arow + 16) * 40 + ak + 16);
    const int bcol = wn * 32 + l15;
    const int bk = lhi * 8;
    v16h fb0 = ldsfrag(sB + bcol * 40 + bk, sB + bcol * 40 + bk + 16);
    v16h fb1 = ldsfrag(sB + (bcol + 16) * 40 + bk, sB + (bcol + 16) * 40 + bk + 16);

    acc00 = wmma32(fa0, fb0, acc00);
    acc01 = wmma32(fa0, fb1, acc01);
    acc10 = wmma32(fa1, fb0, acc10);
    acc11 = wmma32(fa1, fb1, acc11);
  }

  const float bb0 = bvec[bn + wn * 32 + l15];
  const float bb1 = bvec[bn + wn * 32 + 16 + l15];
  __syncthreads();
  {
    const int rb = wm * 32 + lhi * 8;
    const int c0 = wn * 32 + l15;
#pragma unroll
    for (int j = 0; j < 8; ++j) {
      sO[(rb + j) * 72 + c0] = (_Float16)(acc00[j] + bb0);
      sO[(rb + j) * 72 + c0 + 16] = (_Float16)(acc01[j] + bb1);
      sO[(rb + 16 + j) * 72 + c0] = (_Float16)(acc10[j] + bb0);
      sO[(rb + 16 + j) * 72 + c0 + 16] = (_Float16)(acc11[j] + bb1);
    }
  }
  __syncthreads();
  {
    const int r = tid >> 1, sg = tid & 1;
    _Float16* dst = out + (size_t)(bm + r) * 512 + bn + sg * 32;
    const _Float16* src = sO + r * 72 + sg * 32;
#pragma unroll
    for (int q = 0; q < 4; ++q)
      *(v8h*)(dst + q * 8) = *(const v8h*)(src + q * 8);
  }
}

// ============================================================================
// Stage 2: per (window b, head h) attention.  K tile via TDM, V transposed.
// 4 waves, each owns a 16-row query strip (rows 49..63 padded/garbage, masked).
// ============================================================================
__global__ __launch_bounds__(128) void win_attn(
    const _Float16* __restrict__ kv, const float* __restrict__ qg,
    const float* __restrict__ relb, _Float16* __restrict__ ows) {
  __shared__ __align__(16) char smem[24064];
  _Float16* sQ = (_Float16*)smem;            // 64 x 40 halfs
  _Float16* sK = (_Float16*)(smem + 5120);   // 64 x 40 halfs (TDM fills 49 rows)
  _Float16* sV = (_Float16*)(smem + 10240);  // 32(d) x 72 halfs (keys transposed)
  _Float16* sS = (_Float16*)(smem + 14848);  // 64 x 72 halfs (exp'd scores)

  const int tid = threadIdx.x, lane = tid & 31, wave = tid >> 5;
  const int l15 = lane & 15, lhi = lane >> 4;
  const int b = blockIdx.x >> 3, h = blockIdx.x & 7;
  const float scale = 0.1767766952966369f;  // 32^-0.5
  const size_t kvrow0 = (size_t)b * NTOK * 512;

#if USE_TDM
  if (wave == 0) {
    // 2D tile: 16 dwords/row (K head slice), 49 rows, row stride 256 dwords.
    // pad_enable: +4 dwords after every 16 -> LDS row stride 80B.
    unsigned long long ga =
        (unsigned long long)(const void*)kv + (kvrow0 + (size_t)h * HD) * 2ull;
    v4u g0 = {1u, (unsigned)(size_t)sK, (unsigned)(ga & 0xffffffffull),
              (unsigned)((ga >> 32) & 0x1ffffffull) | (2u << 30)};
    v8i g1 = {(int)((2u << 16) | (1u << 20) | (3u << 22) | (3u << 25)),
              (int)(16u << 16),  // tensor_dim0 = 16
              (int)(49u << 16),  // tensor_dim1 = 49
              (int)(16u << 16),  // tile_dim0 = 16
              49,                // tile_dim1 = 49
              256,               // tensor_dim0_stride = 256 dwords
              0, 0};
    v4i gz4 = {0, 0, 0, 0};
    v8i gz8 = {0, 0, 0, 0, 0, 0, 0, 0};
    __builtin_amdgcn_tensor_load_to_lds(g0, g1, gz4, gz4, gz8, 0);
  }
#else
  for (int i = tid; i < NTOK * 16; i += 128) {
    int n = i >> 4, d2 = i & 15;
    *(v2h*)(sK + n * 40 + d2 * 2) =
        *(const v2h*)(kv + kvrow0 + (size_t)n * 512 + h * HD + d2 * 2);
  }
#endif

  // stage Q (scaled, f32 -> f16)
  {
    const float* qp = qg + ((size_t)(b >> 6) * HEADS + h) * (NTOK * HD);
    for (int i = tid; i < NTOK * 16; i += 128) {
      int n = i >> 4, d2 = i & 15;
      float2 f = *(const float2*)(qp + n * HD + d2 * 2);
      v2h hh = {(_Float16)(f.x * scale), (_Float16)(f.y * scale)};
      *(v2h*)(sQ + n * 40 + d2 * 2) = hh;
    }
  }
  // zero V key-tail (49..63), fill V transposed: sV[d][k] (disjoint ranges)
  for (int i = tid; i < 32 * 15; i += 128) {
    int d = i / 15, kk = 49 + (i % 15);
    sV[d * 72 + kk] = (_Float16)0.f;
  }
  for (int i = tid; i < NTOK * HD; i += 128) {
    int n = i >> 5, d = i & 31;
    sV[d * 72 + n] = kv[kvrow0 + (size_t)n * 512 + 256 + h * HD + d];
  }
#if USE_TDM
  if (wave == 0) __builtin_amdgcn_s_wait_tensorcnt(0);
#endif
  __syncthreads();

  // ---- S = Q @ K^T (K-dim = 32, one wmma per 16x16 tile) -------------------
  const int strip = wave * 16;
  const int arow = strip + l15;
  v16h fq = ldsfrag(sQ + arow * 40 + lhi * 8, sQ + arow * 40 + lhi * 8 + 16);

  v8f sAcc[4] = {{}, {}, {}, {}};
#pragma unroll
  for (int nt = 0; nt < 4; ++nt) {
    const _Float16* kp = sK + (nt * 16 + l15) * 40 + lhi * 8;
    v16h fk = ldsfrag(kp, kp + 16);
    sAcc[nt] = wmma32(fq, fk, sAcc[nt]);
  }

  // ---- relative-position bias + pad mask -----------------------------------
#pragma unroll
  for (int j = 0; j < 8; ++j) {
    int qr = strip + lhi * 8 + j;
    int qy = qr / 7, qx = qr - qy * 7;
#pragma unroll
    for (int nt = 0; nt < 4; ++nt) {
      int c = nt * 16 + l15;
      if (qr < NTOK && c < NTOK) {
        int ky = c / 7, kx = c - ky * 7;
        int idx = (qy - ky + 6) * 13 + (qx - kx + 6);
        sAcc[nt][j] += relb[idx * HEADS + h];
      } else {
        sAcc[nt][j] = -1e30f;
      }
    }
  }

  // ---- row softmax (rows live across the 16 lanes of each half-wave) -------
  float rinv[8];
#pragma unroll
  for (int j = 0; j < 8; ++j) {
    float m = fmaxf(fmaxf(sAcc[0][j], sAcc[1][j]), fmaxf(sAcc[2][j], sAcc[3][j]));
    m = fmaxf(m, __shfl_xor(m, 1));
    m = fmaxf(m, __shfl_xor(m, 2));
    m = fmaxf(m, __shfl_xor(m, 4));
    m = fmaxf(m, __shfl_xor(m, 8));
    float t = 0.f;
#pragma unroll
    for (int nt = 0; nt < 4; ++nt) {
      float e = __expf(sAcc[nt][j] - m);
      sAcc[nt][j] = e;
      t += e;
    }
    t += __shfl_xor(t, 1);
    t += __shfl_xor(t, 2);
    t += __shfl_xor(t, 4);
    t += __shfl_xor(t, 8);
    rinv[j] = 1.f / t;
  }

  // ---- C-layout -> A-layout via LDS ----------------------------------------
  {
    _Float16* sSw = sS + strip * 72;
#pragma unroll
    for (int nt = 0; nt < 4; ++nt)
#pragma unroll
      for (int j = 0; j < 8; ++j)
        sSw[(lhi * 8 + j) * 72 + nt * 16 + l15] = (_Float16)sAcc[nt][j];
  }
  __syncthreads();

  // ---- O = S @ V  (K = 64 keys -> 2 wmma steps; N = 32 -> 2 tiles) ---------
  v8f o0 = {}, o1 = {};
#pragma unroll
  for (int kk = 0; kk < 2; ++kk) {
    const _Float16* sp = sS + (strip + l15) * 72 + kk * 32 + lhi * 8;
    v16h fa = ldsfrag(sp, sp + 16);
    const _Float16* vp0 = sV + (l15)*72 + kk * 32 + lhi * 8;
    const _Float16* vp1 = sV + (16 + l15) * 72 + kk * 32 + lhi * 8;
    v16h fb0 = ldsfrag(vp0, vp0 + 16);
    v16h fb1 = ldsfrag(vp1, vp1 + 16);
    o0 = wmma32(fa, fb0, o0);
    o1 = wmma32(fa, fb1, o1);
  }

  // ---- normalize and store f16 [token, h*32+d] -----------------------------
#pragma unroll
  for (int j = 0; j < 8; ++j) {
    int qr = strip + lhi * 8 + j;
    if (qr < NTOK) {
      size_t rowo = ((size_t)b * NTOK + qr) * CDIM + h * HD;
      ows[rowo + l15] = (_Float16)(o0[j] * rinv[j]);
      ows[rowo + 16 + l15] = (_Float16)(o1[j] * rinv[j]);
    }
  }
}

// ============================================================================
// Stage 3: out = o_ws @ proj_w^T + proj_b  (f16 A, f32 out)
// ============================================================================
__global__ __launch_bounds__(256) void proj_gemm(
    const _Float16* __restrict__ a, const float* __restrict__ w,
    const float* __restrict__ bvec, float* __restrict__ out) {
  __shared__ __align__(16) char smem[34816];
  _Float16* sA = (_Float16*)smem;            // 128 x 40 halfs
  _Float16* sB = (_Float16*)(smem + 10240);  // 64 x 40 halfs
  float* sO = (float*)smem;                  // 128 x 68 floats (bounce)

  const int tid = threadIdx.x, lane = tid & 31, wave = tid >> 5;
  const int wm = wave >> 1, wn = wave & 1;
  const int bm = blockIdx.x * 128, bn = blockIdx.y * 64;
  const int l15 = lane & 15, lhi = lane >> 4;

  v8f acc00 = {}, acc01 = {}, acc10 = {}, acc11 = {};

  const int ar = tid >> 1;
  const int as = (tid & 1) * 16;
  const int brn = tid >> 2;
  const int bsg = (tid & 3) * 8;

  for (int k0 = 0; k0 < CDIM; k0 += 32) {
    __syncthreads();
    {
      const _Float16* ap = a + (size_t)(bm + ar) * CDIM + k0 + as;
      *(v8h*)(sA + ar * 40 + as) = *(const v8h*)ap;
      *(v8h*)(sA + ar * 40 + as + 8) = *(const v8h*)(ap + 8);

      const float* wp = w + (size_t)(bn + brn) * CDIM + k0 + bsg;
      float4 b0 = ((const float4*)wp)[0];
      float4 b1 = ((const float4*)wp)[1];
      v8h hb = {(_Float16)b0.x, (_Float16)b0.y, (_Float16)b0.z, (_Float16)b0.w,
                (_Float16)b1.x, (_Float16)b1.y, (_Float16)b1.z, (_Float16)b1.w};
      *(v8h*)(sB + brn * 40 + bsg) = hb;
    }
    __syncthreads();

    const int arow = wm * 32 + l15;
    const int ak = lhi * 8;
    v16h fa0 = ldsfrag(sA + arow * 40 + ak, sA + arow * 40 + ak + 16);
    v16h fa1 = ldsfrag(sA + (arow + 16) * 40 + ak, sA + (arow + 16) * 40 + ak + 16);
    const int bcol = wn * 32 + l15;
    const int bk = lhi * 8;
    v16h fb0 = ldsfrag(sB + bcol * 40 + bk, sB + bcol * 40 + bk + 16);
    v16h fb1 = ldsfrag(sB + (bcol + 16) * 40 + bk, sB + (bcol + 16) * 40 + bk + 16);

    acc00 = wmma32(fa0, fb0, acc00);
    acc01 = wmma32(fa0, fb1, acc01);
    acc10 = wmma32(fa1, fb0, acc10);
    acc11 = wmma32(fa1, fb1, acc11);
  }

  const float bb0 = bvec[bn + wn * 32 + l15];
  const float bb1 = bvec[bn + wn * 32 + 16 + l15];
  __syncthreads();
  {
    const int rb = wm * 32 + lhi * 8;
    const int c0 = wn * 32 + l15;
#pragma unroll
    for (int j = 0; j < 8; ++j) {
      sO[(rb + j) * 68 + c0] = acc00[j] + bb0;
      sO[(rb + j) * 68 + c0 + 16] = acc01[j] + bb1;
      sO[(rb + 16 + j) * 68 + c0] = acc10[j] + bb0;
      sO[(rb + 16 + j) * 68 + c0 + 16] = acc11[j] + bb1;
    }
  }
  __syncthreads();
  {
    const int r = tid >> 1, sg = tid & 1;
    float* dst = out + (size_t)(bm + r) * CDIM + bn + sg * 32;
    const float* src = sO + r * 68 + sg * 32;
#pragma unroll
    for (int q = 0; q < 8; ++q)
      *(float4*)(dst + q * 4) = *(const float4*)(src + q * 4);
  }
}

// ============================================================================
extern "C" void kernel_launch(void* const* d_in, const int* in_sizes, int n_in,
                              void* d_out, int out_size, void* d_ws,
                              size_t ws_size, hipStream_t stream) {
  const float* x = (const float*)d_in[0];
  const float* qg = (const float*)d_in[1];
  const float* qkv_w = (const float*)d_in[2];
  const float* qkv_b = (const float*)d_in[3];
  const float* proj_w = (const float*)d_in[4];
  const float* proj_b = (const float*)d_in[5];
  const float* relb = (const float*)d_in[6];
  float* out = (float*)d_out;

  _Float16* kv_ws = (_Float16*)d_ws;  // [MROWS, 512] f16 (K cols 0..255, V 256..511)
  _Float16* o_ws =
      (_Float16*)((char*)d_ws + (size_t)MROWS * 512 * sizeof(_Float16));  // [MROWS, 256]

  kv_gemm_f16<<<dim3(MROWS / 128, 512 / 64), 256, 0, stream>>>(x, qkv_w, qkv_b,
                                                               kv_ws);
  win_attn<<<dim3(NWIN * HEADS), 128, 0, stream>>>(kv_ws, qg, relb, o_ws);
  proj_gemm<<<dim3(MROWS / 128, CDIM / 64), 256, 0, stream>>>(o_ws, proj_w,
                                                              proj_b, out);
}
